// MultiHeadAttention_50697793962418
// MI455X (gfx1250) — matrix operand order repaired and checked by reading, not compiled, not run on hardware
//
#include <hip/hip_runtime.h>
#include <hip/hip_bf16.h>

// ---------------------------------------------------------------------------
// MHA forward for MI455X (gfx1250): wave32 + v_wmma_f32_16x16x32_f16
//   B=2, T=2048, D=1024, H=16, DK=64
// Register-double-buffered WMMA GEMMs (32x64 tiles) + transpose-free
// flash-style attention (scores^T so exp() feeds WMMA B-operand in-lane).
// ---------------------------------------------------------------------------

typedef __attribute__((ext_vector_type(16))) _Float16 v16h;
typedef __attribute__((ext_vector_type(8)))  _Float16 v8h;
typedef __attribute__((ext_vector_type(8)))  float    v8f;

#define BATCH  2
#define SEQ    2048
#define DMODEL 1024
#define NHEAD  16
#define DK     64
#define MROWS  (BATCH * SEQ)        // 4096

__device__ __forceinline__ v8f wmma_f16(v16h a, v16h b, v8f c) {
    return __builtin_amdgcn_wmma_f32_16x16x32_f16(false, a, false, b,
                                                  (short)0, c, false, false);
}

// ------------------------------ f32 -> f16 ---------------------------------
__global__ void cvt_f32_to_f16_kernel(const float* __restrict__ src,
                                      _Float16* __restrict__ dst, int n) {
    int i = blockIdx.x * blockDim.x + threadIdx.x;
    if (i < n) dst[i] = (_Float16)src[i];
}

// ------------------------- WMMA GEMM: y = A * W^T + b ----------------------
// A: [M, K] f16 row-major.  W: [N, K] f16 row-major (nn.Linear weight).
// Each wave computes a 32x64 tile: 2 A-fragments x 4 W-fragments = 8 WMMA
// per 32-K chunk, with next-chunk fragments prefetched into registers so
// the WMMA pipe overlaps the L2 loads (partial s_wait_loadcnt).
// mode 0: out f32 [M,N]            (final projection, scale must be 1)
// mode 1: out f16 [B,H,T,DK]       (Q with scale=1/8, K with scale=1)
// mode 2: out f16 [B,H,DK,T]       (V transposed for attention stage 2)
__global__ void gemm_wmma_kernel(const _Float16* __restrict__ A,
                                 const _Float16* __restrict__ W,
                                 const float* __restrict__ bias,
                                 float* __restrict__ outF,
                                 _Float16* __restrict__ outH,
                                 int M, int N, int K, int mode, float scale) {
    const int wid  = (blockIdx.x * blockDim.x + threadIdx.x) >> 5;
    const int lane = threadIdx.x & 31;
    const int lm   = lane & 15;     // row/col within 16-tile
    const int lh   = lane >> 4;     // K-half selector

    const int mTiles = M >> 5;            // 32-row tiles
    const int mt = wid % mTiles;
    const int nb = wid / mTiles;          // 64-col block

    const _Float16* A0 = A + (size_t)(mt * 32 + lm) * K + lh * 16;
    const _Float16* A1 = A0 + (size_t)16 * K;
    const _Float16* W0 = W + (size_t)(nb * 64 + lm) * K + lh * 16;
    const _Float16* W1 = W0 + (size_t)16 * K;
    const _Float16* W2 = W0 + (size_t)32 * K;
    const _Float16* W3 = W0 + (size_t)48 * K;

    v8f acc[2][4] = {};

    // prefetch chunk 0
    v16h aN0 = *(const v16h*)(A0);
    v16h aN1 = *(const v16h*)(A1);
    v16h wN0 = *(const v16h*)(W0);
    v16h wN1 = *(const v16h*)(W1);
    v16h wN2 = *(const v16h*)(W2);
    v16h wN3 = *(const v16h*)(W3);

    for (int kc = 0; kc < K; kc += 32) {
        const v16h a0 = aN0, a1 = aN1;
        const v16h w0 = wN0, w1 = wN1, w2 = wN2, w3 = wN3;
        // prefetch next chunk (wraps to 0 on last iter; harmless dummy loads)
        const int kn = (kc + 32 < K) ? kc + 32 : 0;
        aN0 = *(const v16h*)(A0 + kn);
        aN1 = *(const v16h*)(A1 + kn);
        wN0 = *(const v16h*)(W0 + kn);
        wN1 = *(const v16h*)(W1 + kn);
        wN2 = *(const v16h*)(W2 + kn);
        wN3 = *(const v16h*)(W3 + kn);

        acc[0][0] = wmma_f16(a0, w0, acc[0][0]);
        acc[0][1] = wmma_f16(a0, w1, acc[0][1]);
        acc[0][2] = wmma_f16(a0, w2, acc[0][2]);
        acc[0][3] = wmma_f16(a0, w3, acc[0][3]);
        acc[1][0] = wmma_f16(a1, w0, acc[1][0]);
        acc[1][1] = wmma_f16(a1, w1, acc[1][1]);
        acc[1][2] = wmma_f16(a1, w2, acc[1][2]);
        acc[1][3] = wmma_f16(a1, w3, acc[1][3]);
    }

    #pragma unroll
    for (int mi = 0; mi < 2; ++mi) {
        #pragma unroll
        for (int nt = 0; nt < 4; ++nt) {
            const int n = nb * 64 + nt * 16 + lm;
            const float bv = bias[n];
            #pragma unroll
            for (int r = 0; r < 8; ++r) {
                const int m = mt * 32 + mi * 16 + r + 8 * lh;
                const float val = (acc[mi][nt][r] + bv) * scale;
                if (mode == 0) {
                    outF[(size_t)m * N + n] = val;
                } else {
                    const int b  = m >> 11;          // m / SEQ
                    const int t  = m & (SEQ - 1);
                    const int h  = n >> 6;           // n / DK
                    const int dk = n & (DK - 1);
                    if (mode == 1)
                        outH[(((size_t)(b * NHEAD + h)) * SEQ + t) * DK + dk] = (_Float16)val;
                    else
                        outH[(((size_t)(b * NHEAD + h)) * DK + dk) * SEQ + t] = (_Float16)val;
                }
            }
        }
    }
}

// ------------------------------ attention ----------------------------------
// One wave per (b, h, 16-query tile).  Computes S' = K_tile * Q^T so the
// accumulator layout (VGPR<->key, lane<->query) matches the B-operand
// orientation of the second WMMA: exp(S') is packed in-lane, no transpose.
// ctx^T[dk, q] += V^T[dk, keys] * P'[keys, q];  normalize by per-query sum.
// K and V fragments for chunk ch+1 are prefetched during chunk ch.
__global__ void attention_wmma_kernel(const _Float16* __restrict__ Q,
                                      const _Float16* __restrict__ Km,
                                      const _Float16* __restrict__ VT,
                                      _Float16* __restrict__ ctx) {
    const int wid  = (blockIdx.x * blockDim.x + threadIdx.x) >> 5;
    const int lane = threadIdx.x & 31;
    const int lm   = lane & 15;
    const int lh   = lane >> 4;

    const int qt = wid & (SEQ / 16 - 1);       // 128 query tiles
    const int h  = (wid >> 7) & (NHEAD - 1);
    const int b  = wid >> 11;

    const _Float16* Qh = Q  + ((((size_t)(b * NHEAD + h)) * SEQ) + qt * 16) * DK;
    const _Float16* Kh = Km + (((size_t)(b * NHEAD + h)) * SEQ) * DK;
    const _Float16* Vh = VT + (((size_t)(b * NHEAD + h)) * DK) * SEQ;

    // Q as B-operand: lane <-> query column, contiguous dk rows of Q.
    const v16h qb0 = *(const v16h*)(Qh + lm * DK + lh * 16);        // dk 0..31
    const v16h qb1 = *(const v16h*)(Qh + lm * DK + 32 + lh * 16);   // dk 32..63

    v8f   cacc[4] = {};
    float denom = 0.0f;

    // prefetch chunk 0: 2 key tiles (2 dk-halves each) + 4 V^T d-tiles
    v16h kN0 = *(const v16h*)(Kh + (size_t)(lm) * DK + lh * 16);
    v16h kN1 = *(const v16h*)(Kh + (size_t)(lm) * DK + 32 + lh * 16);
    v16h kN2 = *(const v16h*)(Kh + (size_t)(16 + lm) * DK + lh * 16);
    v16h kN3 = *(const v16h*)(Kh + (size_t)(16 + lm) * DK + 32 + lh * 16);
    v16h vN0 = __builtin_shufflevector(*(const v8h*)(Vh + (size_t)(0 * 16 + lm) * SEQ + lh * 8), *(const v8h*)(Vh + (size_t)(0 * 16 + lm) * SEQ + 16 + lh * 8), 0,1,2,3,4,5,6,7,8,9,10,11,12,13,14,15);
    v16h vN1 = __builtin_shufflevector(*(const v8h*)(Vh + (size_t)(1 * 16 + lm) * SEQ + lh * 8), *(const v8h*)(Vh + (size_t)(1 * 16 + lm) * SEQ + 16 + lh * 8), 0,1,2,3,4,5,6,7,8,9,10,11,12,13,14,15);
    v16h vN2 = __builtin_shufflevector(*(const v8h*)(Vh + (size_t)(2 * 16 + lm) * SEQ + lh * 8), *(const v8h*)(Vh + (size_t)(2 * 16 + lm) * SEQ + 16 + lh * 8), 0,1,2,3,4,5,6,7,8,9,10,11,12,13,14,15);
    v16h vN3 = __builtin_shufflevector(*(const v8h*)(Vh + (size_t)(3 * 16 + lm) * SEQ + lh * 8), *(const v8h*)(Vh + (size_t)(3 * 16 + lm) * SEQ + 16 + lh * 8), 0,1,2,3,4,5,6,7,8,9,10,11,12,13,14,15);

    for (int ch = 0; ch < SEQ / 32; ++ch) {
        const v16h k0 = kN0, k1 = kN1, k2 = kN2, k3 = kN3;
        const v16h v0 = vN0, v1 = vN1, v2 = vN2, v3 = vN3;

        // prefetch chunk ch+1 (wraps on last iteration; dummy but valid)
        const int cn = (ch + 1 < SEQ / 32) ? ch + 1 : 0;
        const _Float16* kb = Kh + (size_t)(cn * 32 + lm) * DK + lh * 16;
        kN0 = *(const v16h*)(kb);
        kN1 = *(const v16h*)(kb + 32);
        kN2 = *(const v16h*)(kb + 16 * DK);
        kN3 = *(const v16h*)(kb + 16 * DK + 32);
        const _Float16* vb = Vh + (size_t)lm * SEQ + cn * 32 + lh * 8;
        vN0 = __builtin_shufflevector(*(const v8h*)(vb), *(const v8h*)(vb + 16), 0,1,2,3,4,5,6,7,8,9,10,11,12,13,14,15);
        vN1 = __builtin_shufflevector(*(const v8h*)(vb + (size_t)16 * SEQ), *(const v8h*)(vb + (size_t)16 * SEQ + 16), 0,1,2,3,4,5,6,7,8,9,10,11,12,13,14,15);
        vN2 = __builtin_shufflevector(*(const v8h*)(vb + (size_t)32 * SEQ), *(const v8h*)(vb + (size_t)32 * SEQ + 16), 0,1,2,3,4,5,6,7,8,9,10,11,12,13,14,15);
        vN3 = __builtin_shufflevector(*(const v8h*)(vb + (size_t)48 * SEQ), *(const v8h*)(vb + (size_t)48 * SEQ + 16), 0,1,2,3,4,5,6,7,8,9,10,11,12,13,14,15);

        union { v16h v; _Float16 e[16]; } p;
        // score tile 0 (keys ch*32 .. +15)
        v8f s = {};
        s = wmma_f16(k0, qb0, s);
        s = wmma_f16(k1, qb1, s);
        #pragma unroll
        for (int i = 0; i < 8; ++i) {
            const float e = __expf(s[i]);       // Q pre-scaled by 1/sqrt(dk)
            denom += e;
            p.e[i] = (_Float16)e;
        }
        // score tile 1 (keys ch*32+16 .. +31)
        v8f s2 = {};
        s2 = wmma_f16(k2, qb0, s2);
        s2 = wmma_f16(k3, qb1, s2);
        #pragma unroll
        for (int i = 0; i < 8; ++i) {
            const float e = __expf(s2[i]);
            denom += e;
            p.e[8 + i] = (_Float16)e;
        }

        cacc[0] = wmma_f16(v0, p.v, cacc[0]);
        cacc[1] = wmma_f16(v1, p.v, cacc[1]);
        cacc[2] = wmma_f16(v2, p.v, cacc[2]);
        cacc[3] = wmma_f16(v3, p.v, cacc[3]);
    }

    // per-query softmax denominator: combine the two key-row halves
    const float inv = 1.0f / (denom + __shfl_xor(denom, 16, 32));

    const int q = qt * 16 + lm;
    _Float16* crow = ctx + ((size_t)(b * SEQ + q)) * DMODEL + h * DK;
    #pragma unroll
    for (int dt = 0; dt < 4; ++dt) {
        v8h o;
        #pragma unroll
        for (int r = 0; r < 8; ++r)
            o[r] = (_Float16)(cacc[dt][r] * inv);
        *(v8h*)(crow + dt * 16 + 8 * lh) = o;   // 16B aligned b128 store
    }
}

// ------------------------------- launcher ----------------------------------
extern "C" void kernel_launch(void* const* d_in, const int* in_sizes, int n_in,
                              void* d_out, int out_size, void* d_ws, size_t ws_size,
                              hipStream_t stream) {
    const float* x  = (const float*)d_in[0];
    const float* Wq = (const float*)d_in[1];
    const float* bq = (const float*)d_in[2];
    const float* Wk = (const float*)d_in[3];
    const float* bk = (const float*)d_in[4];
    const float* Wv = (const float*)d_in[5];
    const float* bv = (const float*)d_in[6];
    const float* Wo = (const float*)d_in[7];
    const float* bo = (const float*)d_in[8];
    float* out = (float*)d_out;

    const size_t nX = (size_t)MROWS * DMODEL;           // 4 Mi elements
    const size_t nW = (size_t)DMODEL * DMODEL;          // 1 Mi elements
    const size_t nP = (size_t)BATCH * NHEAD * SEQ * DK; // 4 Mi elements

    char* ws = (char*)d_ws;
    _Float16* xh  = (_Float16*)ws;                  ws += nX * sizeof(_Float16);
    _Float16* Wqh = (_Float16*)ws;                  ws += nW * sizeof(_Float16);
    _Float16* Wkh = (_Float16*)ws;                  ws += nW * sizeof(_Float16);
    _Float16* Wvh = (_Float16*)ws;                  ws += nW * sizeof(_Float16);
    _Float16* Woh = (_Float16*)ws;                  ws += nW * sizeof(_Float16);
    _Float16* Qb  = (_Float16*)ws;                  ws += nP * sizeof(_Float16);
    _Float16* Kb  = (_Float16*)ws;                  ws += nP * sizeof(_Float16);
    _Float16* VT  = (_Float16*)ws;                  ws += nP * sizeof(_Float16);
    _Float16* ctx = (_Float16*)ws;                  ws += nX * sizeof(_Float16);

    // 1) convert inputs to f16
    cvt_f32_to_f16_kernel<<<(int)((nX + 255) / 256), 256, 0, stream>>>(x,  xh,  (int)nX);
    cvt_f32_to_f16_kernel<<<(int)((nW + 255) / 256), 256, 0, stream>>>(Wq, Wqh, (int)nW);
    cvt_f32_to_f16_kernel<<<(int)((nW + 255) / 256), 256, 0, stream>>>(Wk, Wkh, (int)nW);
    cvt_f32_to_f16_kernel<<<(int)((nW + 255) / 256), 256, 0, stream>>>(Wv, Wvh, (int)nW);
    cvt_f32_to_f16_kernel<<<(int)((nW + 255) / 256), 256, 0, stream>>>(Wo, Woh, (int)nW);

    // 2) Q/K/V projections: waves = (M/32)*(N/64) = 2048 -> 256 blocks of 256
    const int gemmBlocks = (MROWS / 32) * (DMODEL / 64) * 32 / 256;
    const float qscale = 0.125f;   // 1/sqrt(DK)
    gemm_wmma_kernel<<<gemmBlocks, 256, 0, stream>>>(xh, Wqh, bq, nullptr, Qb,
        MROWS, DMODEL, DMODEL, 1, qscale);
    gemm_wmma_kernel<<<gemmBlocks, 256, 0, stream>>>(xh, Wkh, bk, nullptr, Kb,
        MROWS, DMODEL, DMODEL, 1, 1.0f);
    gemm_wmma_kernel<<<gemmBlocks, 256, 0, stream>>>(xh, Wvh, bv, nullptr, VT,
        MROWS, DMODEL, DMODEL, 2, 1.0f);

    // 3) attention: one wave per (b, h, 16-query tile) = 4096 waves
    const int attnBlocks = BATCH * NHEAD * (SEQ / 16) * 32 / 256;
    attention_wmma_kernel<<<attnBlocks, 256, 0, stream>>>(Qb, Kb, VT, ctx);

    // 4) output projection -> f32 d_out
    gemm_wmma_kernel<<<gemmBlocks, 256, 0, stream>>>(ctx, Woh, bo, out, nullptr,
        MROWS, DMODEL, DMODEL, 0, 1.0f);

    (void)in_sizes; (void)n_in; (void)out_size; (void)ws_size;
}
